// NeuralNetworkSimplified_5368709120128
// MI455X (gfx1250) — matrix operand order repaired and checked by reading, not compiled, not run on hardware
//
#include <hip/hip_runtime.h>

typedef __attribute__((ext_vector_type(4)))  _Float16 v4h;
typedef __attribute__((ext_vector_type(8)))  _Float16 v8h;
typedef __attribute__((ext_vector_type(16))) _Float16 v16h;
typedef __attribute__((ext_vector_type(8)))  float    v8f;

// ---------------- tiny prep kernels ----------------

__global__ void binarize_w_kernel(const float* __restrict__ W, _Float16* __restrict__ Wo, int n) {
    int i = blockIdx.x * blockDim.x + threadIdx.x;
    if (i >= n) return;
    float w = W[i];
    Wo[i] = (_Float16)((w > 0.f) ? 1.f : ((w < 0.f) ? -1.f : 0.f));
}

__global__ void sign_f32_kernel(const float* __restrict__ b, float* __restrict__ bo, int n) {
    int i = blockIdx.x * blockDim.x + threadIdx.x;
    if (i >= n) return;
    float w = b[i];
    bo[i] = (w > 0.f) ? 1.f : ((w < 0.f) ? -1.f : 0.f);
}

// ---------------- WMMA GEMM: C[BM x BN] = A[BM x K] * Bw[BN x K]^T + bias ----------------
// BM=64, BK=32, 256 threads = 8 waves in a 2(M) x 4(N) grid.
// Each wave: TM=2 x TN tiles of 16x16 (f32 acc). One block spans full N width.

#define GEMM_BM   64
#define GEMM_BK   32
#define GEMM_LDK  48   // BK + 16 halves pad: 96B row stride, keeps b128 LDS reads 16B-aligned

__device__ inline v16h frag_from(const v8h lo, const v8h hi) {
    v16h t;
#pragma unroll
    for (int i = 0; i < 8; ++i) { t[i] = lo[i]; t[i + 8] = hi[i]; }
    return t;
}

// A 16x32 f16 frag: lanes 0-15 row M=lane, K 0-7 (vgpr0-3) + 16-23 (vgpr4-7);
//                   lanes 16-31 same rows, K 8-15 + 24-31.
__device__ inline v16h load_frag_a(const _Float16* lds, int rowBase, int lane) {
    const _Float16* p = lds + (rowBase + (lane & 15)) * GEMM_LDK + ((lane >> 4) << 3);
    v8h lo = *(const v8h*)p;
    v8h hi = *(const v8h*)(p + 16);
    return frag_from(lo, hi);
}

// B 32x16 f16 frag: lanes 0-15 col N=lane, K 0-7 + 16-23; lanes 16-31 K 8-15 + 24-31.
__device__ inline v16h load_frag_b(const _Float16* lds, int rowBase, int lane) {
    const _Float16* p = lds + (rowBase + (lane & 15)) * GEMM_LDK + ((lane >> 4) << 3);
    v8h lo = *(const v8h*)p;
    v8h hi = *(const v8h*)(p + 16);
    return frag_from(lo, hi);
}

template <typename AT, int BN, int TN>
__global__ void __launch_bounds__(256)
gemm_bin_kernel(const AT* __restrict__ A, const _Float16* __restrict__ Bw,
                const float* __restrict__ bias, float* __restrict__ C, int K) {
    __shared__ _Float16 ldsA[GEMM_BM * GEMM_LDK];
    __shared__ _Float16 ldsB[BN * GEMM_LDK];

    const int tid   = threadIdx.x;
    const int lane  = tid & 31;
    const int wave  = tid >> 5;
    const int waveM = wave & 1;   // 0..1
    const int waveN = wave >> 1;  // 0..3
    const int blockRow = blockIdx.x * GEMM_BM;

    v8f acc[2][TN];
#pragma unroll
    for (int tm = 0; tm < 2; ++tm)
#pragma unroll
        for (int tn = 0; tn < TN; ++tn) acc[tm][tn] = (v8f)0.f;

    constexpr int CA = GEMM_BM * GEMM_BK / (256 * 4);  // 2
    constexpr int CB = BN * GEMM_BK / (256 * 4);       // 8 (BN=256) or 4 (BN=128)

    for (int k0 = 0; k0 < K; k0 += GEMM_BK) {
        __syncthreads();
        // ---- A tile -> LDS (f32 source converted to f16 on the fly) ----
#pragma unroll
        for (int c = 0; c < CA; ++c) {
            int linear = (c * 256 + tid) * 4;
            int row = linear / GEMM_BK, kk = linear % GEMM_BK;
            const AT* src = A + (size_t)(blockRow + row) * K + k0 + kk;
            v4h d;
            if constexpr (sizeof(AT) == 4) {
                float4 f = *(const float4*)src;
                d[0] = (_Float16)f.x; d[1] = (_Float16)f.y;
                d[2] = (_Float16)f.z; d[3] = (_Float16)f.w;
            } else {
                d = *(const v4h*)src;
            }
            *(v4h*)&ldsA[row * GEMM_LDK + kk] = d;
        }
        // ---- B tile (pre-binarized f16 weights, row-major [N][K]) -> LDS ----
#pragma unroll
        for (int c = 0; c < CB; ++c) {
            int linear = (c * 256 + tid) * 4;
            int row = linear / GEMM_BK, kk = linear % GEMM_BK;
            *(v4h*)&ldsB[row * GEMM_LDK + kk] =
                *(const v4h*)(Bw + (size_t)row * K + k0 + kk);
        }
        __syncthreads();

        v16h af[2], bf[TN];
#pragma unroll
        for (int tm = 0; tm < 2; ++tm)
            af[tm] = load_frag_a(ldsA, waveM * 32 + tm * 16, lane);
#pragma unroll
        for (int tn = 0; tn < TN; ++tn)
            bf[tn] = load_frag_b(ldsB, waveN * TN * 16 + tn * 16, lane);

#pragma unroll
        for (int tm = 0; tm < 2; ++tm)
#pragma unroll
            for (int tn = 0; tn < TN; ++tn)
                acc[tm][tn] = __builtin_amdgcn_wmma_f32_16x16x32_f16(
                    false, af[tm], false, bf[tn], (short)0, acc[tm][tn], false, false);
    }

    // ---- epilogue: C/D layout (lane n = col, vgpr r = row r + 8*(lane>=16)) ----
#pragma unroll
    for (int tm = 0; tm < 2; ++tm) {
#pragma unroll
        for (int tn = 0; tn < TN; ++tn) {
            int col  = waveN * TN * 16 + tn * 16 + (lane & 15);
            int row0 = blockRow + waveM * 32 + tm * 16 + ((lane >> 4) << 3);
            float bs = bias[col];
#pragma unroll
            for (int r = 0; r < 8; ++r)
                C[(size_t)(row0 + r) * BN + col] = acc[tm][tn][r] + bs;
        }
    }
}

// ---------------- deterministic per-column sum / sumsq (two-stage) ----------------

__global__ void colstats_partial_kernel(const float* __restrict__ H, int rowsPerChunk,
                                        int cols, float* __restrict__ partial) {
    int col   = threadIdx.x;
    int chunk = blockIdx.x;
    if (col >= cols) return;
    const float* p = H + (size_t)chunk * rowsPerChunk * cols + col;
    float s = 0.f, s2 = 0.f;
    for (int r = 0; r < rowsPerChunk; ++r) {
        float v = p[(size_t)r * cols];
        s += v; s2 += v * v;
    }
    partial[chunk * 2 * cols + col]        = s;
    partial[chunk * 2 * cols + cols + col] = s2;
}

__global__ void colstats_reduce_kernel(const float* __restrict__ partial, int chunks,
                                       int cols, float* __restrict__ stats) {
    int col = threadIdx.x;
    if (col >= cols) return;
    float s = 0.f, s2 = 0.f;
    for (int c = 0; c < chunks; ++c) {
        s  += partial[c * 2 * cols + col];
        s2 += partial[c * 2 * cols + cols + col];
    }
    stats[col]        = s;
    stats[cols + col] = s2;
}

// ---------------- BN(train, biased var) + hardtanh, f32 -> f16 ----------------

__global__ void bn_act_kernel(const float* __restrict__ H, const float* __restrict__ stats,
                              const float* __restrict__ g, const float* __restrict__ be,
                              _Float16* __restrict__ out, int rows, int cols) {
    int idx = blockIdx.x * blockDim.x + threadIdx.x;
    if (idx >= rows * cols) return;
    int col = idx % cols;
    float invN = 1.f / (float)rows;
    float mu  = stats[col] * invN;
    float var = stats[cols + col] * invN - mu * mu;
    float inv = rsqrtf(var + 1e-5f);
    float v = (H[idx] - mu) * inv * g[col] + be[col];
    v = fminf(1.f, fmaxf(-1.f, v));
    out[idx] = (_Float16)v;
}

// ---------------- final float linear [B,128] x [12,128]^T + b4 ----------------

__global__ void final_linear_kernel(const _Float16* __restrict__ H, const float* __restrict__ W4,
                                    const float* __restrict__ b4, float* __restrict__ out,
                                    int rows) {
    int idx = blockIdx.x * blockDim.x + threadIdx.x;
    if (idx >= rows * 12) return;
    int b = idx / 12, c = idx % 12;
    const _Float16* h = H + (size_t)b * 128;
    const float* w = W4 + c * 128;
    float acc = b4[c];
#pragma unroll 8
    for (int k = 0; k < 128; ++k) acc += (float)h[k] * w[k];
    out[idx] = acc;
}

// ---------------- launch ----------------

extern "C" void kernel_launch(void* const* d_in, const int* in_sizes, int n_in,
                              void* d_out, int out_size, void* d_ws, size_t ws_size,
                              hipStream_t stream) {
    const float* x   = (const float*)d_in[0];
    const float* W1  = (const float*)d_in[1];
    const float* b1  = (const float*)d_in[2];
    const float* g1  = (const float*)d_in[3];
    const float* be1 = (const float*)d_in[4];
    const float* W2  = (const float*)d_in[5];
    const float* b2  = (const float*)d_in[6];
    const float* g2  = (const float*)d_in[7];
    const float* be2 = (const float*)d_in[8];
    const float* W4  = (const float*)d_in[9];
    const float* b4  = (const float*)d_in[10];
    // d_in[11] = y, unused by forward pass

    constexpr int B = 16384, IN = 4096, N1 = 256, N2 = 128, NC = 12;
    constexpr int CHUNKS = 16;

    char* ws = (char*)d_ws;
    size_t off = 0;
    auto alloc = [&](size_t bytes) -> void* {
        void* p = ws + off;
        off = (off + bytes + 255) & ~(size_t)255;
        return p;
    };
    _Float16* Wb1   = (_Float16*)alloc((size_t)N1 * IN * 2);
    float*    sb1   = (float*)   alloc((size_t)N1 * 4);
    _Float16* Wb2   = (_Float16*)alloc((size_t)N2 * N1 * 2);
    float*    sb2   = (float*)   alloc((size_t)N2 * 4);
    float*    H1    = (float*)   alloc((size_t)B * N1 * 4);
    _Float16* H1c   = (_Float16*)alloc((size_t)B * N1 * 2);
    float*    H2    = (float*)   alloc((size_t)B * N2 * 4);
    _Float16* H2c   = (_Float16*)alloc((size_t)B * N2 * 2);
    float*    part  = (float*)   alloc((size_t)CHUNKS * 2 * N1 * 4);
    float*    st1   = (float*)   alloc((size_t)2 * N1 * 4);
    float*    st2   = (float*)   alloc((size_t)2 * N2 * 4);

    // weight binarization
    binarize_w_kernel<<<(N1 * IN + 255) / 256, 256, 0, stream>>>(W1, Wb1, N1 * IN);
    sign_f32_kernel<<<1, 256, 0, stream>>>(b1, sb1, N1);
    binarize_w_kernel<<<(N2 * N1 + 255) / 256, 256, 0, stream>>>(W2, Wb2, N2 * N1);
    sign_f32_kernel<<<1, 256, 0, stream>>>(b2, sb2, N2);

    // layer 1: x [B,4096] (f32, converted to f16 in-tile) @ sign(W1)^T + sign(b1)
    gemm_bin_kernel<float, N1, 4><<<B / GEMM_BM, 256, 0, stream>>>(x, Wb1, sb1, H1, IN);
    colstats_partial_kernel<<<CHUNKS, 256, 0, stream>>>(H1, B / CHUNKS, N1, part);
    colstats_reduce_kernel<<<1, 256, 0, stream>>>(part, CHUNKS, N1, st1);
    bn_act_kernel<<<(B * N1) / 256, 256, 0, stream>>>(H1, st1, g1, be1, H1c, B, N1);

    // layer 2
    gemm_bin_kernel<_Float16, N2, 2><<<B / GEMM_BM, 256, 0, stream>>>(H1c, Wb2, sb2, H2, N1);
    colstats_partial_kernel<<<CHUNKS, 256, 0, stream>>>(H2, B / CHUNKS, N2, part);
    colstats_reduce_kernel<<<1, 256, 0, stream>>>(part, CHUNKS, N2, st2);
    bn_act_kernel<<<(B * N2) / 256, 256, 0, stream>>>(H2, st2, g2, be2, H2c, B, N2);

    // final float linear -> d_out [B,12] f32
    final_linear_kernel<<<(B * NC + 255) / 256, 256, 0, stream>>>(H2c, W4, b4, (float*)d_out, B);
}
